// MaskedVideoTransformer_72559177498923
// MI455X (gfx1250) — matrix operand order repaired and checked by reading, not compiled, not run on hardware
//
#include <hip/hip_runtime.h>
#include <hip/hip_bf16.h>

// ---------------------------------------------------------------------------
// Masked video transformer (MAE) forward for MI455X / gfx1250 (wave32, WMMA).
//   B=16, F=10, C=3, IMG=128, P=16  -> NP=640 patches, patch dim 768
//   D=128, E=32 -> DM=160, L=12 layers per stack, H=16 heads (dh=10), FF=2048
// Mixed precision: f16 activations + f16 weights, f32 WMMA accumulation,
// f32 LayerNorm/softmax statistics. All GEMMs on v_wmma_f32_16x16x32_f16.
// ---------------------------------------------------------------------------

typedef __attribute__((ext_vector_type(16))) _Float16 v16h;
typedef __attribute__((ext_vector_type(8)))  _Float16 v8h;
typedef __attribute__((ext_vector_type(8)))  float    v8f;

#define NP    640
#define BATCH 16
#define ROWS  (BATCH * NP)   // 10240
#define DM    160
#define DD    128
#define EE    32
#define FF    2048
#define HQ    16
#define DH    10
#define PDIM  768            // C*P*P
#define NLAYER 12
#define QTILES 40            // ceil(640/16)

// ---- WMMA fragment index helpers (contiguous A/B k order used by this file) --------
// A frag (16x32 f16): lane m = lane&15; half p maps to
//   k = p + 16*(lane>=16)  -> two contiguous 8-half (16B) runs.
// B frag (32x16 f16): lane n = lane&15; half p -> k = p + 16*(lane>=16).
// C frag: vgpr r -> m = r + 8*(lane>=16), n = lane&15.
__device__ __forceinline__ int a_kflat(int p, int hi) {
  return p + hi * 16;
}

__device__ __forceinline__ v16h load_afrag(const _Float16* __restrict__ Arow,
                                           int hi) {
  v8h lo = *(const v8h*)(Arow + hi * 16);       // k = hi*16 .. hi*16+7
  v8h hh = *(const v8h*)(Arow + 8 + hi * 16);   // k = 8+hi*16 .. 15+hi*16
  return __builtin_shufflevector(lo, hh, 0, 1, 2, 3, 4, 5, 6, 7, 8, 9, 10, 11,
                                 12, 13, 14, 15);
}

__device__ __forceinline__ v8f wmma_f16(v16h a, v16h b, v8f c) {
  return __builtin_amdgcn_wmma_f32_16x16x32_f16(false, a, false, b, (short)0,
                                                c, false, false);
}

// ---------------------------------------------------------------------------
// Weight repack: W[N][K] f32 (row major, B = W^T) -> fragment-native f16.
// Group g = kt*(N/16)+nt holds 32 lanes x 16 halves; lane writes 32B coalesced.
// ---------------------------------------------------------------------------
__global__ __launch_bounds__(256) void pack_weight_f16(
    const float* __restrict__ W, _Float16* __restrict__ out, int N, int K) {
  int g    = blockIdx.x * (blockDim.x >> 5) + (threadIdx.x >> 5);
  int lane = threadIdx.x & 31;
  int tN   = N >> 4;
  int total = (K >> 5) * tN;
  if (g >= total) return;
  int kt = g / tN;
  int nt = g % tN;
  int hi = (lane >> 4) & 1;
  int n  = nt * 16 + (lane & 15);
  _Float16* o = out + ((size_t)g * 32 + lane) * 16;
#pragma unroll
  for (int p = 0; p < 16; ++p) {
    int k = kt * 32 + p + hi * 16;
    o[p] = (_Float16)W[(size_t)n * K + k];
  }
}

// ---------------------------------------------------------------------------
// GEMM: C[M,N](f16) = A[M,K](f16, lda) @ Bpacked(f16) + bias(f32), opt. ReLU.
// One wave computes a 32x32 strip: 2 A frags x 2 B frags -> 4 WMMAs / k-step
// (B reused across M-tiles, A loads amortized; weights stay L2-resident).
// M % 32 == 0, N % 32 == 0, K % 32 == 0.
// ---------------------------------------------------------------------------
__global__ __launch_bounds__(256) void gemm_f16(
    const _Float16* __restrict__ A, int lda,
    const _Float16* __restrict__ Bp,
    const float* __restrict__ bias,
    _Float16* __restrict__ C, int ldc,
    int M, int N, int K, int relu) {
  const int lane  = threadIdx.x & 31;
  const int wave  = blockIdx.x * (blockDim.x >> 5) + (threadIdx.x >> 5);
  const int tN    = N >> 5;
  const int strips = (M >> 5) * tN;
  if (wave >= strips) return;
  const int mi = wave / tN, nj = wave % tN;
  const int m0 = mi << 5, n0 = nj << 5;
  const int hi   = (lane >> 4) & 1;
  const int nlo  = lane & 15;
  const int tN16 = N >> 4;

  const _Float16* A0 = A + (size_t)(m0 + nlo) * lda;
  const _Float16* A1 = A0 + (size_t)16 * lda;

  v8f acc00 = {}, acc01 = {}, acc10 = {}, acc11 = {};
  const int ksteps = K >> 5;
  for (int kt = 0; kt < ksteps; ++kt) {
    v16h a0 = load_afrag(A0 + (kt << 5), hi);
    v16h a1 = load_afrag(A1 + (kt << 5), hi);
    const _Float16* bp =
        Bp + ((size_t)(kt * tN16 + (nj << 1)) * 32 + lane) * 16;
    v16h b0 = *(const v16h*)bp;
    v16h b1 = *(const v16h*)(bp + 32 * 16);
    acc00 = wmma_f16(a0, b0, acc00);
    acc01 = wmma_f16(a0, b1, acc01);
    acc10 = wmma_f16(a1, b0, acc10);
    acc11 = wmma_f16(a1, b1, acc11);
  }
  const float bz0 = bias ? bias[n0 + nlo] : 0.f;
  const float bz1 = bias ? bias[n0 + nlo + 16] : 0.f;
#pragma unroll
  for (int r = 0; r < 8; ++r) {
    int m = m0 + r + hi * 8;
    int n = n0 + nlo;
    float v00 = acc00[r] + bz0, v01 = acc01[r] + bz1;
    float v10 = acc10[r] + bz0, v11 = acc11[r] + bz1;
    if (relu) {
      v00 = fmaxf(v00, 0.f); v01 = fmaxf(v01, 0.f);
      v10 = fmaxf(v10, 0.f); v11 = fmaxf(v11, 0.f);
    }
    C[(size_t)m * ldc + n]             = (_Float16)v00;
    C[(size_t)m * ldc + n + 16]        = (_Float16)v01;
    C[(size_t)(m + 16) * ldc + n]      = (_Float16)v10;
    C[(size_t)(m + 16) * ldc + n + 16] = (_Float16)v11;
  }
}

// ---------------------------------------------------------------------------
// Attention: one wave per (batch, head, 16-query block). Scores kept in LDS
// (16 x 640 f32 strip, 41KB of the 320KB WGP LDS). dh=10 zero-padded to K=32.
// eff = nkeep for encoder (device value), 640 for decoder.
// ---------------------------------------------------------------------------
__global__ __launch_bounds__(32) void attention_kernel(
    const _Float16* __restrict__ qkv, _Float16* __restrict__ obuf,
    const int* __restrict__ nkeepPtr, int fixedS) {
  const int lane = threadIdx.x;
  const int qb = blockIdx.x % QTILES;
  const int h  = (blockIdx.x / QTILES) & (HQ - 1);
  const int b  = blockIdx.x / (QTILES * HQ);
  const int eff = (fixedS > 0) ? fixedS : *nkeepPtr;
  const int s0 = qb * 16;
  if (s0 >= eff) return;

  __shared__ float sc[16][648];
  __shared__ float rowsum[16];

  const int hi  = (lane >> 4) & 1;
  const int nlo = lane & 15;
  const float qscale = 0.316227766016838f;  // 1/sqrt(dh)

  // Q fragment (scaled), zero-padded past dh
  v16h qa;
  {
    const _Float16* qrow = qkv + (size_t)(b * NP + s0 + nlo) * 480 + h * DH;
#pragma unroll
    for (int p = 0; p < 16; ++p) {
      int k = a_kflat(p, hi);
      qa[p] = (k < DH) ? (_Float16)((float)qrow[k] * qscale) : (_Float16)0.f;
    }
  }

  // scores = Q @ K^T, one WMMA per 16-key tile
  const int ktiles = (eff + 15) >> 4;
  for (int t = 0; t < ktiles; ++t) {
    v16h kb;
    const _Float16* krow =
        qkv + (size_t)(b * NP + t * 16 + nlo) * 480 + DM + h * DH;
#pragma unroll
    for (int p = 0; p < 16; ++p) {
      int d = p + hi * 16;
      kb[p] = (d < DH) ? krow[d] : (_Float16)0.f;
    }
    v8f s = {};
    s = wmma_f16(qa, kb, s);
#pragma unroll
    for (int r = 0; r < 8; ++r) sc[r + hi * 8][t * 16 + nlo] = s[r];
  }
  __syncthreads();

  // mask invalid keys
  for (int r = 0; r < 16; ++r)
    for (int c = eff + lane; c < NP; c += 32) sc[r][c] = -1e30f;
  __syncthreads();

  // softmax: unnormalized exp in LDS, per-row sums via wave32 shuffles
  for (int r = 0; r < 16; ++r) {
    float mx = -1e30f;
    for (int c = lane; c < NP; c += 32) mx = fmaxf(mx, sc[r][c]);
#pragma unroll
    for (int o = 16; o >= 1; o >>= 1) mx = fmaxf(mx, __shfl_xor(mx, o, 32));
    float sum = 0.f;
    for (int c = lane; c < NP; c += 32) {
      float e = __expf(sc[r][c] - mx);
      sc[r][c] = e;
      sum += e;
    }
#pragma unroll
    for (int o = 16; o >= 1; o >>= 1) sum += __shfl_xor(sum, o, 32);
    if (lane == 0) rowsum[r] = sum;
  }
  __syncthreads();

  // O = P @ V, streamed over 32-key chunks
  v8f acc = {};
  for (int t = 0; t < NP / 32; ++t) {
    v16h pa, vb;
#pragma unroll
    for (int p = 0; p < 16; ++p)
      pa[p] = (_Float16)sc[nlo][t * 32 + a_kflat(p, hi)];
#pragma unroll
    for (int p = 0; p < 16; ++p) {
      int key = t * 32 + p + hi * 16;
      vb[p] = (nlo < DH)
                  ? qkv[(size_t)(b * NP + key) * 480 + 2 * DM + h * DH + nlo]
                  : (_Float16)0.f;
    }
    acc = wmma_f16(pa, vb, acc);
  }
  if (nlo < DH) {
#pragma unroll
    for (int r = 0; r < 8; ++r) {
      int m = r + hi * 8;
      obuf[(size_t)(b * NP + s0 + m) * DM + h * DH + nlo] =
          (_Float16)(acc[r] / rowsum[m]);
    }
  }
}

// ---------------------------------------------------------------------------
// Residual + LayerNorm (f32 stats): h = LN(h + t) * w + b. Wave per token.
// ---------------------------------------------------------------------------
__global__ __launch_bounds__(256) void resln_kernel(
    _Float16* __restrict__ h, const _Float16* __restrict__ t,
    const float* __restrict__ w, const float* __restrict__ bb, int rows) {
  int wave = blockIdx.x * (blockDim.x >> 5) + (threadIdx.x >> 5);
  int lane = threadIdx.x & 31;
  if (wave >= rows) return;
  float x[5];
  float s = 0.f;
#pragma unroll
  for (int i = 0; i < 5; ++i) {
    int c = lane + i * 32;
    x[i] = (float)h[(size_t)wave * DM + c] + (float)t[(size_t)wave * DM + c];
    s += x[i];
  }
#pragma unroll
  for (int o = 16; o >= 1; o >>= 1) s += __shfl_xor(s, o, 32);
  float mean = s * (1.f / DM);
  float vs = 0.f;
#pragma unroll
  for (int i = 0; i < 5; ++i) { float d = x[i] - mean; vs += d * d; }
#pragma unroll
  for (int o = 16; o >= 1; o >>= 1) vs += __shfl_xor(vs, o, 32);
  float inv = rsqrtf(vs * (1.f / DM) + 1e-5f);
#pragma unroll
  for (int i = 0; i < 5; ++i) {
    int c = lane + i * 32;
    h[(size_t)wave * DM + c] = (_Float16)((x[i] - mean) * inv * w[c] + bb[c]);
  }
}

// ---------------------------------------------------------------------------
// Data-movement kernels
// ---------------------------------------------------------------------------
__global__ void patchify_kernel(const float* __restrict__ X,
                                _Float16* __restrict__ pf) {
  size_t i = (size_t)blockIdx.x * blockDim.x + threadIdx.x;
  if (i >= (size_t)ROWS * PDIM) return;
  int col = (int)(i % PDIM);
  size_t row = i / PDIM;
  int n = (int)(row % NP);
  int b = (int)(row / NP);
  int c = col >> 8, py = (col >> 4) & 15, px = col & 15;
  int f = n >> 6, gy = (n >> 3) & 7, gx = n & 7;
  int y = gy * 16 + py, x = gx * 16 + px;
  pf[i] = (_Float16)X[(((size_t)(b * 10 + f) * 3 + c) * 128 + y) * 128 + x];
}

__global__ void embfill_kernel(_Float16* __restrict__ Xc,
                               const float* __restrict__ Emb) {
  int i = blockIdx.x * blockDim.x + threadIdx.x;
  if (i >= ROWS * EE) return;
  int e = i & 31;
  int row = i >> 5;
  int n = row % NP;
  Xc[(size_t)row * DM + DD + e] = (_Float16)Emb[n * EE + e];
}

__global__ void build_idx_kernel(const int* __restrict__ keep, int* kidx,
                                 int* rank, int* nkeep) {
  if (threadIdx.x == 0 && blockIdx.x == 0) {
    int c = 0;
    for (int n = 0; n < NP; ++n) {
      if (keep[n] != 0) { kidx[c] = n; rank[n] = c; ++c; }
      else rank[n] = -1;
    }
    *nkeep = c;
  }
}

__global__ void gather_kernel(const _Float16* __restrict__ Xc,
                              _Float16* __restrict__ encH,
                              const int* __restrict__ kidx,
                              const int* __restrict__ nkeep) {
  size_t i = (size_t)blockIdx.x * blockDim.x + threadIdx.x;
  if (i >= (size_t)ROWS * DM) return;
  int d = (int)(i % DM);
  size_t row = i / DM;
  int j = (int)(row % NP);
  int b = (int)(row / NP);
  _Float16 v = (_Float16)0.f;
  if (j < *nkeep) v = Xc[(size_t)(b * NP + kidx[j]) * DM + d];
  encH[i] = v;
}

__global__ void scatter_kernel(const _Float16* __restrict__ encH,
                               _Float16* __restrict__ decH,
                               const int* __restrict__ rank,
                               const float* __restrict__ MissEmb,
                               const float* __restrict__ Emb) {
  size_t i = (size_t)blockIdx.x * blockDim.x + threadIdx.x;
  if (i >= (size_t)ROWS * DM) return;
  int d = (int)(i % DM);
  size_t row = i / DM;
  int n = (int)(row % NP);
  int b = (int)(row / NP);
  int r = rank[n];
  _Float16 v;
  if (r >= 0) v = encH[(size_t)(b * NP + r) * DM + d];
  else v = (_Float16)((d < DD) ? MissEmb[n * DD + d] : Emb[n * EE + (d - DD)]);
  decH[i] = v;
}

__global__ void unpatchify_kernel(const _Float16* __restrict__ rec,
                                  float* __restrict__ out) {
  size_t i = (size_t)blockIdx.x * blockDim.x + threadIdx.x;
  if (i >= (size_t)BATCH * 10 * 3 * 128 * 128) return;
  int x = (int)(i & 127);
  int y = (int)((i >> 7) & 127);
  int c = (int)((i / (128 * 128)) % 3);
  int f = (int)((i / (128 * 128 * 3)) % 10);
  int b = (int)(i / (128 * 128 * 30));
  int gy = y >> 4, py = y & 15, gx = x >> 4, px = x & 15;
  int n = f * 64 + gy * 8 + gx;
  out[i] = (float)rec[(size_t)(b * NP + n) * PDIM + (c << 8) + (py << 4) + px];
}

__global__ void blind_kernel(const float* __restrict__ X,
                             const int* __restrict__ keep,
                             float* __restrict__ out) {
  size_t i = (size_t)blockIdx.x * blockDim.x + threadIdx.x;
  if (i >= (size_t)BATCH * 10 * 3 * 128 * 128) return;
  int x = (int)(i & 127);
  int y = (int)((i >> 7) & 127);
  int f = (int)((i / (128 * 128 * 3)) % 10);
  int gy = y >> 4, gx = x >> 4;
  int n = f * 64 + gy * 8 + gx;
  out[i] = keep[n] ? X[i] : 1.0f;
}

// ---------------------------------------------------------------------------
// Host launcher
// ---------------------------------------------------------------------------
extern "C" void kernel_launch(void* const* d_in, const int* in_sizes, int n_in,
                              void* d_out, int out_size, void* d_ws,
                              size_t ws_size, hipStream_t stream) {
  (void)in_sizes; (void)n_in; (void)out_size;

  const float* X       = (const float*)d_in[0];
  const int*   keep    = (const int*)d_in[1];
  const float* P_w     = (const float*)d_in[2];
  const float* P_b     = (const float*)d_in[3];
  const float* Pinv_w  = (const float*)d_in[4];
  const float* Pinv_b  = (const float*)d_in[5];
  const float* Emb     = (const float*)d_in[6];
  const float* MissEmb = (const float*)d_in[7];

  // Workspace carve-up (256B aligned), all activations f16
  char* base = (char*)d_ws;
  size_t off = 0;
  auto alloc = [&](size_t bytes) -> void* {
    void* p = base + off;
    off += (bytes + 255) & ~(size_t)255;
    return p;
  };
  _Float16* pf   = (_Float16*)alloc((size_t)ROWS * PDIM * 2);  // reused: rec
  _Float16* Xc   = (_Float16*)alloc((size_t)ROWS * DM * 2);
  _Float16* encH = (_Float16*)alloc((size_t)ROWS * DM * 2);
  _Float16* decH = (_Float16*)alloc((size_t)ROWS * DM * 2);
  _Float16* qkvB = (_Float16*)alloc((size_t)ROWS * 480 * 2);
  _Float16* obuf = (_Float16*)alloc((size_t)ROWS * DM * 2);
  _Float16* tmp  = (_Float16*)alloc((size_t)ROWS * DM * 2);
  _Float16* ff1  = (_Float16*)alloc((size_t)ROWS * FF * 2);
  _Float16* packBuf = (_Float16*)alloc((size_t)DM * FF * 2);  // max pack size
  int* kidx  = (int*)alloc(NP * 4);
  int* rank  = (int*)alloc(NP * 4);
  int* nkeep = (int*)alloc(4);
  if (off > ws_size) return;  // workspace too small; bail safely

  auto launch_pack = [&](const float* W, int N, int K) {
    int groups = (K / 32) * (N / 16);
    int blocks = (groups + 7) / 8;
    pack_weight_f16<<<blocks, 256, 0, stream>>>(W, packBuf, N, K);
  };
  auto launch_gemm = [&](const _Float16* A, int lda, const float* bias,
                         _Float16* C, int ldc, int M, int N, int K, int relu) {
    int strips = (M / 32) * (N / 32);
    int blocks = (strips + 7) / 8;
    gemm_f16<<<blocks, 256, 0, stream>>>(A, lda, packBuf, bias, C, ldc, M, N,
                                         K, relu);
  };

  // 0) keep-mask -> indices / ranks / count (device-side, graph-safe)
  build_idx_kernel<<<1, 32, 0, stream>>>(keep, kidx, rank, nkeep);

  // 1) patchify + patch projection + positional-embedding concat -> Xc
  {
    size_t tot = (size_t)ROWS * PDIM;
    patchify_kernel<<<(int)((tot + 255) / 256), 256, 0, stream>>>(X, pf);
    launch_pack(P_w, DD, PDIM);
    launch_gemm(pf, PDIM, P_b, Xc, DM, ROWS, DD, PDIM, 0);
    embfill_kernel<<<(ROWS * EE + 255) / 256, 256, 0, stream>>>(Xc, Emb);
  }

  // 2) gather kept tokens (zero-fill padding rows)
  {
    size_t tot = (size_t)ROWS * DM;
    gather_kernel<<<(int)((tot + 255) / 256), 256, 0, stream>>>(Xc, encH, kidx,
                                                                nkeep);
  }

  // Transformer stack runner
  auto run_stack = [&](_Float16* h, int pbase, int fixedS) {
    const float* in_w  = (const float*)d_in[pbase + 0];
    const float* in_b  = (const float*)d_in[pbase + 1];
    const float* out_w = (const float*)d_in[pbase + 2];
    const float* out_b = (const float*)d_in[pbase + 3];
    const float* l1_w  = (const float*)d_in[pbase + 4];
    const float* l1_b  = (const float*)d_in[pbase + 5];
    const float* l2_w  = (const float*)d_in[pbase + 6];
    const float* l2_b  = (const float*)d_in[pbase + 7];
    const float* ln1w  = (const float*)d_in[pbase + 8];
    const float* ln1b  = (const float*)d_in[pbase + 9];
    const float* ln2w  = (const float*)d_in[pbase + 10];
    const float* ln2b  = (const float*)d_in[pbase + 11];
    for (int l = 0; l < NLAYER; ++l) {
      // qkv projection
      launch_pack(in_w + (size_t)l * 3 * DM * DM, 3 * DM, DM);
      launch_gemm(h, DM, in_b + (size_t)l * 3 * DM, qkvB, 3 * DM, ROWS, 3 * DM,
                  DM, 0);
      // attention (wave per b,h,16-query block)
      attention_kernel<<<BATCH * HQ * QTILES, 32, 0, stream>>>(qkvB, obuf,
                                                               nkeep, fixedS);
      // output projection + residual + LN1
      launch_pack(out_w + (size_t)l * DM * DM, DM, DM);
      launch_gemm(obuf, DM, out_b + (size_t)l * DM, tmp, DM, ROWS, DM, DM, 0);
      resln_kernel<<<ROWS / 8, 256, 0, stream>>>(h, tmp, ln1w + l * DM,
                                                 ln1b + l * DM, ROWS);
      // FFN: relu(h @ W1^T + b1) @ W2^T + b2, residual + LN2
      launch_pack(l1_w + (size_t)l * FF * DM, FF, DM);
      launch_gemm(h, DM, l1_b + (size_t)l * FF, ff1, FF, ROWS, FF, DM, 1);
      launch_pack(l2_w + (size_t)l * DM * FF, DM, FF);
      launch_gemm(ff1, FF, l2_b + (size_t)l * DM, tmp, DM, ROWS, DM, FF, 0);
      resln_kernel<<<ROWS / 8, 256, 0, stream>>>(h, tmp, ln2w + l * DM,
                                                 ln2b + l * DM, ROWS);
    }
  };

  // 3) encoder on kept tokens (seq length = nkeep, resolved on device)
  run_stack(encH, 8, /*fixedS=*/0);

  // 4) scatter into decoder sequence (missing-token + pos emb elsewhere)
  {
    size_t tot = (size_t)ROWS * DM;
    scatter_kernel<<<(int)((tot + 255) / 256), 256, 0, stream>>>(
        encH, decH, rank, MissEmb, Emb);
  }

  // 5) decoder on full sequence
  run_stack(decH, 20, /*fixedS=*/NP);

  // 6) un-projection (first 128 dims) + unpatchify -> output[0]
  launch_pack(Pinv_w, PDIM, DD);
  launch_gemm(decH, DM, Pinv_b, pf /*rec*/, PDIM, ROWS, PDIM, DD, 0);
  {
    size_t tot = (size_t)BATCH * 10 * 3 * 128 * 128;
    unpatchify_kernel<<<(int)((tot + 255) / 256), 256, 0, stream>>>(
        pf, (float*)d_out);
    // 7) blind reconstruction -> output[1]
    blind_kernel<<<(int)((tot + 255) / 256), 256, 0, stream>>>(
        X, keep, (float*)d_out + tot);
  }
}
